// StructuredElmanAttentionCell_79130477462077
// MI455X (gfx1250) — matrix operand order repaired and checked by reading, not compiled, not run on hardware
//
#include <hip/hip_runtime.h>
#include <hip/hip_bf16.h>

typedef __attribute__((ext_vector_type(16))) _Float16 v16h;
typedef __attribute__((ext_vector_type(8)))  _Float16 v8h;
typedef __attribute__((ext_vector_type(8)))  float    v8f;

#define T_STEPS 512
#define BATCH   8
#define NHEADS  16
#define NSTATE  32
#define HDIM    64
#define MRANK   8
#define DATTN   32
#define D_INNER (NHEADS * HDIM)      // 1024
#define SCALE_F 0.17677669529663687f // 32^-0.5

__device__ __forceinline__ float fast_rcp(float x) {
  return __builtin_amdgcn_rcpf(x);   // v_rcp_f32
}
__device__ __forceinline__ float silu_f(float x) {
  return x * fast_rcp(1.0f + __expf(-x));
}

// A-fragment (16x32 f16) from row-major LDS [M][ld], tile row m0.
// lane: m = m0+(lane&15), half = lane>>4.
// elems 0..7  -> K = half*8 + e      ; elems 8..15 -> K = half*8 + 16 + e
__device__ __forceinline__ v16h load_a_frag(const _Float16* src, int ld, int m0) {
  const int lane = threadIdx.x & 31;
  const _Float16* p = src + (m0 + (lane & 15)) * ld + (lane >> 4) * 8;
  v8h c0 = *(const v8h*)(p);
  v8h c1 = *(const v8h*)(p + 16);
  return __builtin_shufflevector(c0, c1, 0,1,2,3,4,5,6,7,8,9,10,11,12,13,14,15);
}

// B-fragment (32x16 f16) from TRANSPOSED row-major LDS [N][ld] (ld = K stride).
// lane: n = n0+(lane&15); elems 0..7 -> K = half*8 + e ; elems 8..15 -> K = half*8 + 16 + e
__device__ __forceinline__ v16h load_b_fragT(const _Float16* src, int ld, int n0) {
  const int lane = threadIdx.x & 31;
  const _Float16* p = src + (n0 + (lane & 15)) * ld + (lane >> 4) * 8;
  v8h c0 = *(const v8h*)(p);
  v8h c1 = *(const v8h*)(p + 16);
  return __builtin_shufflevector(c0, c1, 0,1,2,3,4,5,6,7,8,9,10,11,12,13,14,15);
}

__device__ __forceinline__ v8f wmma_f16(v16h a, v16h b, v8f c) {
  return __builtin_amdgcn_wmma_f32_16x16x32_f16(
      false, a, false, b, (short)0, c, false, false);
}

__global__ __launch_bounds__(256)
void elman_attn_scan(const float* __restrict__ Bp, const float* __restrict__ Xp,
                     const float* __restrict__ aRaw, const float* __restrict__ Z,
                     const float* __restrict__ H0, const float* __restrict__ aBias,
                     const float* __restrict__ Wq, const float* __restrict__ Wk,
                     const float* __restrict__ Wv, const float* __restrict__ Wo,
                     float* __restrict__ Yout, float* __restrict__ Hout)
{
  // Staging in LDS; recurrent state H lives in registers (8 f32/lane).
  __shared__ __align__(16) _Float16 sHf[NSTATE * HDIM];    // f16 H mirror (A operand)
  __shared__ __align__(16) _Float16 wqT[DATTN * HDIM];     // W_q^T [kd][p]
  __shared__ __align__(16) _Float16 wkT[DATTN * HDIM];     // W_k^T
  __shared__ __align__(16) _Float16 wvT[DATTN * HDIM];     // W_v^T
  __shared__ __align__(16) _Float16 woT[HDIM * DATTN];     // W_o^T [p][kd]
  __shared__ __align__(16) _Float16 sB[NSTATE * 32];       // update A [n][r], K pad 32
  __shared__ __align__(16) _Float16 sX[HDIM * 32];         // update B^T [p][r], K pad 32
  __shared__ __align__(16) _Float16 sQ[NSTATE * DATTN];    // Q  [n][kd] (A operand)
  __shared__ __align__(16) _Float16 sK[NSTATE * DATTN];    // K  [m][kd] (B^T for scores)
  __shared__ __align__(16) _Float16 sVT[DATTN * NSTATE];   // V^T[kd][m] (B^T for attnV)
  __shared__ __align__(16) _Float16 sAttn[NSTATE * NSTATE];// attn [n][m] (A operand)
  __shared__ __align__(16) _Float16 sAV[NSTATE * DATTN];   // attnV [n][kd] (A operand)
  __shared__ float sS[NSTATE * NSTATE];
  __shared__ float sY[HDIM];
  __shared__ float sAlpha;

  const int tid  = threadIdx.x;
  const int lane = tid & 31;
  const int w    = tid >> 5;       // wave id 0..7 (wave32)
  const int half = lane >> 4;
  const int bh   = blockIdx.x;     // b*NHEADS + h
  const int b    = bh / NHEADS;
  const int h    = bh % NHEADS;

  const int mt_u = w >> 2;         // 2x4 grid of 16x16 tiles over 32x64
  const int nt_u = w & 3;
  const int colu = nt_u * 16 + (lane & 15);   // this lane's fixed H column
  const int row0 = mt_u * 16 + 8 * half;      // rows row0..row0+7 (v index)

  // ---- one-time preload ----
  float hreg[8];                               // H[row0+v][colu], whole scan
#pragma unroll
  for (int v = 0; v < 8; ++v)
    hreg[v] = H0[(size_t)bh * NSTATE * HDIM + (row0 + v) * HDIM + colu];

  for (int i = tid; i < HDIM * DATTN; i += 256) {
    int p  = i >> 5, kd = i & 31;          // natural W_{q,k,v}[p][kd]
    wqT[kd * HDIM + p] = (_Float16)Wq[(size_t)h * HDIM * DATTN + i];
    wkT[kd * HDIM + p] = (_Float16)Wk[(size_t)h * HDIM * DATTN + i];
    wvT[kd * HDIM + p] = (_Float16)Wv[(size_t)h * HDIM * DATTN + i];
    int kd2 = i >> 6, p2 = i & 63;         // natural W_o[kd][p]
    woT[p2 * DATTN + kd2] = (_Float16)Wo[(size_t)h * DATTN * HDIM + i];
  }
  for (int i = tid; i < NSTATE * 32; i += 256) sB[i] = (_Float16)0.f;
  for (int i = tid; i < HDIM * 32;   i += 256) sX[i] = (_Float16)0.f;
  const float abias = aBias[h];
  __syncthreads();

  for (int t = 0; t < T_STEPS; ++t) {
    const long tb = (long)t * BATCH + b;
    const float* bsrc = Bp + ((tb * NHEADS) + h) * (NSTATE * MRANK);
    const float* xsrc = Xp + ((tb * NHEADS) + h) * (HDIM * MRANK);

    // ---- phase 1: stage B_t [n][r], X_t [p][r] as f16; alpha; zero sY ----
    {
      sB[(tid >> 3) * 32 + (tid & 7)] = (_Float16)bsrc[tid];   // 256 == NS*MR
      for (int j = tid; j < HDIM * MRANK; j += 256)
        sX[(j >> 3) * 32 + (j & 7)] = (_Float16)xsrc[j];       // natural layout
      if (tid < HDIM) sY[tid] = 0.f;
      if (tid == 0) {
        float x  = aRaw[tb * NHEADS + h] + abias;
        float sp = (x > 20.f) ? x : log1pf(__expf(x));         // softplus
        sAlpha = fast_rcp(1.0f + __expf(sp));                  // sigmoid(-softplus)
      }
      if (t + 1 < T_STEPS) {  // prefetch next step's tiles into near caches
        const float* nbB = bsrc + (size_t)BATCH * NHEADS * NSTATE * MRANK;
        const float* nbX = xsrc + (size_t)BATCH * NHEADS * HDIM * MRANK;
        if (tid < 8)  __builtin_prefetch(nbB + tid * 32, 0, 3);
        if (tid < 16) __builtin_prefetch(nbX + tid * 32, 0, 3);
      }
    }
    __syncthreads();

    // ---- phase 2: H = silu(alpha*H + B_t @ X_t^T), all in registers ----
    {
      v16h a  = load_a_frag(sB, 32, mt_u * 16);
      v16h bb = load_b_fragT(sX, 32, nt_u * 16);
      v8f  c  = {};
      c = wmma_f16(a, bb, c);
      const float alpha = sAlpha;
#pragma unroll
      for (int v = 0; v < 8; ++v)
        hreg[v] = silu_f(alpha * hreg[v] + c[v]);
    }

    // ---- phase 3: periodic state attention (residual) ----
    if (((t + 1) & 7) == 0) {
      // publish f16 H mirror from registers
#pragma unroll
      for (int v = 0; v < 8; ++v)
        sHf[(row0 + v) * HDIM + colu] = (_Float16)hreg[v];
      __syncthreads();

      // Q/K/V = H(32x64) @ W(64x32): 12 tiles, K=64 (2 k-steps)
      for (int tau = w; tau < 12; tau += 8) {
        int mat = tau >> 2, sub = tau & 3, mt = sub >> 1, nt = sub & 1;
        const _Float16* WT = (mat == 0) ? wqT : (mat == 1) ? wkT : wvT;
        v8f c = {};
#pragma unroll
        for (int kk = 0; kk < 2; ++kk) {
          v16h a  = load_a_frag(sHf + kk * 32, HDIM, mt * 16);
          v16h bb = load_b_fragT(WT + kk * 32, HDIM, nt * 16);
          c = wmma_f16(a, bb, c);
        }
        int col = nt * 16 + (lane & 15);
#pragma unroll
        for (int v = 0; v < 8; ++v) {
          int row = mt * 16 + v + 8 * half;
          _Float16 hv = (_Float16)c[v];
          if      (mat == 0) sQ[row * DATTN + col]  = hv;   // [n][kd]
          else if (mat == 1) sK[row * DATTN + col]  = hv;   // [m][kd] natural
          else               sVT[col * NSTATE + row] = hv;  // [kd][m] transposed
        }
      }
      __syncthreads();

      // S = (Q @ K^T) * scale: 32x32, K=32, waves 0..3
      if (w < 4) {
        int mt = w >> 1, nt = w & 1;
        v16h a  = load_a_frag(sQ, DATTN, mt * 16);
        v16h bb = load_b_fragT(sK, DATTN, nt * 16);
        v8f  c  = {};
        c = wmma_f16(a, bb, c);
        int col = nt * 16 + (lane & 15);
#pragma unroll
        for (int v = 0; v < 8; ++v)
          sS[(mt * 16 + v + 8 * half) * NSTATE + col] = c[v] * SCALE_F;
      }
      __syncthreads();

      // row softmax -> f16 probabilities
      if (tid < NSTATE) {
        const float* r = sS + tid * NSTATE;
        float m = r[0];
        for (int j = 1; j < NSTATE; ++j) m = fmaxf(m, r[j]);
        float s = 0.f;
        for (int j = 0; j < NSTATE; ++j) s += __expf(r[j] - m);
        float inv = fast_rcp(s);
        for (int j = 0; j < NSTATE; ++j)
          sAttn[tid * NSTATE + j] = (_Float16)(__expf(r[j] - m) * inv);
      }
      __syncthreads();

      // AV = attn @ V: 32x32, K=32, waves 0..3
      if (w < 4) {
        int mt = w >> 1, nt = w & 1;
        v16h a  = load_a_frag(sAttn, NSTATE, mt * 16);
        v16h bb = load_b_fragT(sVT, NSTATE, nt * 16);
        v8f  c  = {};
        c = wmma_f16(a, bb, c);
        int col = nt * 16 + (lane & 15);
#pragma unroll
        for (int v = 0; v < 8; ++v)
          sAV[(mt * 16 + v + 8 * half) * DATTN + col] = (_Float16)c[v];
      }
      __syncthreads();

      // H += AV @ W_o: 32x64, K=32, all 8 waves, accumulate into registers
      {
        v16h a  = load_a_frag(sAV, DATTN, mt_u * 16);
        v16h bb = load_b_fragT(woT, DATTN, nt_u * 16);
        v8f  c  = {};
        c = wmma_f16(a, bb, c);
#pragma unroll
        for (int v = 0; v < 8; ++v)
          hreg[v] += c[v];
      }
    }

    // ---- phase 4: y = H.sum(rows) * silu(z), row-sums from registers ----
    {
      float psum = 0.f;
#pragma unroll
      for (int v = 0; v < 8; ++v) psum += hreg[v];
      atomicAdd(&sY[colu], psum);            // ds_add_f32 (4 adds per column)
    }
    __syncthreads();
    if (tid < HDIM) {
      const float zz = Z[tb * D_INNER + h * HDIM + tid];
      Yout[tb * D_INNER + h * HDIM + tid] = sY[tid] * silu_f(zz);
    }
    __syncthreads();
  }

  // ---- H_final straight from registers ----
#pragma unroll
  for (int v = 0; v < 8; ++v)
    Hout[(size_t)bh * NSTATE * HDIM + (row0 + v) * HDIM + colu] = hreg[v];
}

extern "C" void kernel_launch(void* const* d_in, const int* in_sizes, int n_in,
                              void* d_out, int out_size, void* d_ws, size_t ws_size,
                              hipStream_t stream) {
  (void)in_sizes; (void)n_in; (void)d_ws; (void)ws_size; (void)out_size;
  const float* Bp    = (const float*)d_in[0];
  const float* Xp    = (const float*)d_in[1];
  const float* aRaw  = (const float*)d_in[2];
  const float* Z     = (const float*)d_in[3];
  const float* H0    = (const float*)d_in[4];
  const float* aBias = (const float*)d_in[5];
  const float* Wq    = (const float*)d_in[6];
  const float* Wk    = (const float*)d_in[7];
  const float* Wv    = (const float*)d_in[8];
  const float* Wo    = (const float*)d_in[9];

  float* Yout = (float*)d_out;                                  // [T,B,D_INNER]
  float* Hout = Yout + (size_t)T_STEPS * BATCH * D_INNER;       // [B,NH,NS,HD]

  dim3 grid(BATCH * NHEADS);   // 128 persistent channel workgroups
  dim3 block(256);             // 8 wave32 waves
  elman_attn_scan<<<grid, block, 0, stream>>>(Bp, Xp, aRaw, Z, H0, aBias,
                                              Wq, Wk, Wv, Wo, Yout, Hout);
}
